// PointINet_65120294142111
// MI455X (gfx1250) — compile-verified, matrix operand order checked by reading, not run on hardware
//
#include <hip/hip_runtime.h>

typedef __bf16 bf16;
typedef __attribute__((ext_vector_type(16))) __bf16 v16bf;
typedef __attribute__((ext_vector_type(8)))  __bf16 v8bf;
typedef __attribute__((ext_vector_type(8)))  float  v8f;

#define DEV static __device__ __forceinline__

DEV v8f wmma_bf16(v16bf a, v16bf b, v8f c) {
    return __builtin_amdgcn_wmma_f32_16x16x32_bf16(false, a, false, b, (short)0, c, false, false);
}

// Build a 16-element bf16 fragment from two contiguous 8-element (16B) loads.
DEV v16bf ld2x8(const bf16* lo_p, const bf16* hi_p) {
    v8bf lo = *(const v8bf*)lo_p;
    v8bf hi = *(const v8bf*)hi_p;
    v16bf r;
#pragma unroll
    for (int i = 0; i < 8; ++i) { r[i] = lo[i]; r[8 + i] = hi[i]; }
    return r;
}

// ---------------------------------------------------------------------------
// Weight pre-pack: W[K,N] f32 -> Wp bf16 in B-fragment order, zero padded to
// Kpad x Npad.  Per (kTile, n): 32 bf16 ordered [k0..7 | k16..23 | k8..15 |
// k24..31]; lane n reads 16 contiguous bf16 at offset half*16.
// ---------------------------------------------------------------------------
__global__ void pin_packw_kernel(const float* __restrict__ W, bf16* __restrict__ Wp,
                                 int K, int N, int Npad, int total)
{
    int tid = blockIdx.x * blockDim.x + threadIdx.x;
    if (tid >= total) return;
    int e  = tid & 31;
    int n  = (tid >> 5) % Npad;
    int kt = tid / (32 * Npad);
    int sub = e >> 3, o = e & 7;
    int k = kt * 32 + ((sub == 0) ? o : (sub == 1) ? 16 + o : (sub == 2) ? 8 + o : 24 + o);
    float v = (k < K && n < N) ? W[(size_t)k * N + n] : 0.f;
    Wp[tid] = (bf16)v;
}

// ---------------------------------------------------------------------------
// GEMM: C[M,N] = act( A[M,Kpad](bf16, zero-padded) * Wpacked + bias[N] )
// Template MTxNT tiles per wave; zero branches in the K loop.
// <2,4>: one wave computes a 32x64 strip (8 WMMA / K-step, B reused 2x).
// <1,1>: narrow heads (Npad == 16).
// ---------------------------------------------------------------------------
template <int MT, int NT>
__global__ void pin_gemmT_kernel(const bf16* __restrict__ A, int lda,
                                 const bf16* __restrict__ Wp,
                                 const float* __restrict__ bias,
                                 void* __restrict__ Cout, int ldc,
                                 int M, int N, int Npad, int K,
                                 int relu, int out_bf16)
{
    const int wave = threadIdx.x >> 5;
    const int lane = threadIdx.x & 31;
    const int mBase = (blockIdx.y * 4 + wave) * (MT * 16);
    if (mBase >= M) return;                       // wave-uniform
    const int nBase0 = blockIdx.x * (NT * 16);

    const int half  = lane >> 4;
    const int idx16 = lane & 15;

    v8f acc[MT][NT];
#pragma unroll
    for (int i = 0; i < MT; ++i)
#pragma unroll
        for (int j = 0; j < NT; ++j)
            acc[i][j] = (v8f){0.f,0.f,0.f,0.f,0.f,0.f,0.f,0.f};

    const bf16* ap[MT];
#pragma unroll
    for (int i = 0; i < MT; ++i)
        ap[i] = A + (size_t)(mBase + i * 16 + idx16) * lda + half * 8;
    const bf16* wp = Wp + ((size_t)(nBase0 + idx16)) * 32 + half * 16;
    const size_t wstep = (size_t)Npad * 32;

    for (int k0 = 0; k0 < K; k0 += 32) {
        v16bf a[MT];
#pragma unroll
        for (int i = 0; i < MT; ++i) { a[i] = ld2x8(ap[i], ap[i] + 16); ap[i] += 32; }
#pragma unroll
        for (int j = 0; j < NT; ++j) {
            v16bf b = ld2x8(wp + j * 512, wp + j * 512 + 8);
#pragma unroll
            for (int i = 0; i < MT; ++i) acc[i][j] = wmma_bf16(a[i], b, acc[i][j]);
        }
        wp += wstep;
    }

#pragma unroll
    for (int j = 0; j < NT; ++j) {
        int n = nBase0 + j * 16 + idx16;
        float bv = (n < N) ? bias[n] : 0.f;
#pragma unroll
        for (int i = 0; i < MT; ++i) {
#pragma unroll
            for (int r = 0; r < 8; ++r) {
                int row = mBase + i * 16 + (half ? 8 + r : r);
                float v = acc[i][j][r] + bv;
                if (relu) v = v > 0.f ? v : 0.f;
                if (row < M && n < N) {
                    if (out_bf16) ((bf16*)Cout)[(size_t)row * ldc + n] = (bf16)v;
                    else          ((float*)Cout)[(size_t)row * ldc + n] = v;
                }
            }
        }
    }
}

// ---------------------------------------------------------------------------
// Farthest point sampling: one block per batch element.
// ---------------------------------------------------------------------------
#define FPS_T 256
__global__ void pin_fps_kernel(const float* __restrict__ xyz, int xs,
                               int Npts, int npoint, int* __restrict__ idx_out)
{
    const int b = blockIdx.x;
    const float* X = xyz + (size_t)b * Npts * xs;
    int* out = idx_out + (size_t)b * npoint;

    __shared__ float dist[2048];
    __shared__ float rv[FPS_T];
    __shared__ int   ri[FPS_T];
    __shared__ int   s_far;

    for (int i = threadIdx.x; i < Npts; i += blockDim.x) dist[i] = 1e10f;
    if (threadIdx.x == 0) s_far = 0;
    __syncthreads();

    for (int it = 0; it < npoint; ++it) {
        const int far = s_far;
        if (threadIdx.x == 0) out[it] = far;
        const float cx = X[(size_t)far * xs + 0];
        const float cy = X[(size_t)far * xs + 1];
        const float cz = X[(size_t)far * xs + 2];
        float bestv = -1.f; int besti = 0;
        for (int i = threadIdx.x; i < Npts; i += blockDim.x) {
            float dx = X[(size_t)i * xs + 0] - cx;
            float dy = X[(size_t)i * xs + 1] - cy;
            float dz = X[(size_t)i * xs + 2] - cz;
            float d  = dx*dx + dy*dy + dz*dz;
            float dd = fminf(dist[i], d);
            dist[i] = dd;
            if (dd > bestv) { bestv = dd; besti = i; }
        }
        rv[threadIdx.x] = bestv; ri[threadIdx.x] = besti;
        __syncthreads();
        for (int s = FPS_T / 2; s > 0; s >>= 1) {
            if (threadIdx.x < s) {
                float ov = rv[threadIdx.x + s]; int oi = ri[threadIdx.x + s];
                if (ov > rv[threadIdx.x] ||
                    (ov == rv[threadIdx.x] && oi < ri[threadIdx.x])) {
                    rv[threadIdx.x] = ov; ri[threadIdx.x] = oi;
                }
            }
            __syncthreads();
        }
        if (threadIdx.x == 0) s_far = ri[0];
        __syncthreads();
    }
}

__global__ void pin_gather_xyz_kernel(const float* __restrict__ xyz, int xs,
                                      const int* __restrict__ idx,
                                      float* __restrict__ nx,
                                      int S, int Npts, int total)
{
    int tid = blockIdx.x * blockDim.x + threadIdx.x;
    if (tid >= total) return;
    int b = tid / S;
    int gi = idx[tid];
    const float* p = xyz + ((size_t)b * Npts + gi) * xs;
    float* o = nx + (size_t)tid * 3;
    o[0] = p[0]; o[1] = p[1]; o[2] = p[2];
}

__global__ void pin_ball_query_kernel(const float* __restrict__ xyz, int xs,
                                      const float* __restrict__ new_xyz,
                                      int Npts, int S, float r2, int nsample,
                                      int* __restrict__ gidx, int total)
{
    int tid = blockIdx.x * blockDim.x + threadIdx.x;
    if (tid >= total) return;
    int b = tid / S;
    const float* X = xyz + (size_t)b * Npts * xs;
    const float* c = new_xyz + (size_t)tid * 3;
    int* g = gidx + (size_t)tid * nsample;
    int cnt = 0;
    for (int i = 0; i < Npts && cnt < nsample; ++i) {
        float dx = X[(size_t)i * xs + 0] - c[0];
        float dy = X[(size_t)i * xs + 1] - c[1];
        float dz = X[(size_t)i * xs + 2] - c[2];
        if (dx*dx + dy*dy + dz*dz <= r2) g[cnt++] = i;
    }
    for (; cnt < nsample; ++cnt) g[cnt] = 0;   // reference pads with index 0
}

// grouped = [centered_xyz(3) | points(Cpts) | 0-pad] -> bf16 rows, lda = Kpad
__global__ void pin_sa_gather_kernel(const float* __restrict__ xyz, int xs,
                                     const float* __restrict__ pts, int ps,
                                     const float* __restrict__ new_xyz,
                                     const int* __restrict__ gidx,
                                     bf16* __restrict__ Aout,
                                     int S, int nsample, int Npts, int Cpts,
                                     int Kpad, int total)
{
    int tid = blockIdx.x * blockDim.x + threadIdx.x;
    if (tid >= total) return;
    int bs = tid / nsample;
    int b  = bs / S;
    int gi = gidx[tid];
    const float* p = xyz + ((size_t)b * Npts + gi) * xs;
    const float* c = new_xyz + (size_t)bs * 3;
    bf16* row = Aout + (size_t)tid * Kpad;
    row[0] = (bf16)(p[0] - c[0]);
    row[1] = (bf16)(p[1] - c[1]);
    row[2] = (bf16)(p[2] - c[2]);
    const float* f = pts + ((size_t)b * Npts + gi) * ps;
    for (int ch = 0; ch < Cpts; ++ch) row[3 + ch] = (bf16)f[ch];
    for (int ch = 3 + Cpts; ch < Kpad; ++ch) row[ch] = (bf16)0.f;
}

__global__ void pin_maxpool_kernel(const float* __restrict__ act,
                                   float* __restrict__ pooled,
                                   int groups, int nsample, int C, int total)
{
    int tid = blockIdx.x * blockDim.x + threadIdx.x;
    if (tid >= total) return;
    int g = tid / C, c = tid % C;
    const float* a = act + (size_t)g * nsample * C + c;
    float m = a[0];
    for (int j = 1; j < nsample; ++j) m = fmaxf(m, a[(size_t)j * C]);
    pooled[(size_t)g * C + c] = m;
}

// Feature propagation: 3-NN inverse-distance interp (broadcast when m2==1),
// writes concat row [pts1(C1) | interp(C2) | 0-pad] as bf16, lda = Kpad.
__global__ void pin_fp_concat_kernel(const float* __restrict__ xyz1, int x1s,
                                     const float* __restrict__ xyz2,
                                     const float* __restrict__ pts1, int p1s, int C1,
                                     const float* __restrict__ pts2, int C2,
                                     bf16* __restrict__ Aout, int Kpad,
                                     int n1, int m2, int total)
{
    int tid = blockIdx.x * blockDim.x + threadIdx.x;
    if (tid >= total) return;
    int b = tid / n1, i = tid % n1;
    bf16* row = Aout + (size_t)tid * Kpad;
    const float* p1 = pts1 + ((size_t)b * n1 + i) * p1s;
    for (int c = 0; c < C1; ++c) row[c] = (bf16)p1[c];

    if (m2 == 1) {
        const float* p2 = pts2 + (size_t)b * C2;
        for (int c = 0; c < C2; ++c) row[C1 + c] = (bf16)p2[c];
    } else {
        const float* q  = xyz1 + ((size_t)b * n1 + i) * x1s;
        const float* X2 = xyz2 + (size_t)b * m2 * 3;
        float d0 = 1e30f, d1 = 1e30f, d2 = 1e30f;
        int   i0 = 0, i1 = 0, i2 = 0;
        for (int j = 0; j < m2; ++j) {
            float dx = q[0] - X2[j*3+0];
            float dy = q[1] - X2[j*3+1];
            float dz = q[2] - X2[j*3+2];
            float d  = dx*dx + dy*dy + dz*dz;
            if (d < d0)      { d2=d1;i2=i1; d1=d0;i1=i0; d0=d;i0=j; }
            else if (d < d1) { d2=d1;i2=i1; d1=d;i1=j; }
            else if (d < d2) { d2=d;i2=j; }
        }
        float w0 = 1.f / fmaxf(d0, 1e-10f);
        float w1 = 1.f / fmaxf(d1, 1e-10f);
        float w2 = 1.f / fmaxf(d2, 1e-10f);
        float ws = w0 + w1 + w2;
        w0 /= ws; w1 /= ws; w2 /= ws;
        const float* P2 = pts2 + (size_t)b * m2 * C2;
        for (int c = 0; c < C2; ++c)
            row[C1 + c] = (bf16)(w0 * P2[(size_t)i0*C2+c] +
                                 w1 * P2[(size_t)i1*C2+c] +
                                 w2 * P2[(size_t)i2*C2+c]);
    }
    for (int c = C1 + C2; c < Kpad; ++c) row[c] = (bf16)0.f;
}

// comb = concat(points1, points2) along token axis, f32 -> bf16
__global__ void pin_comb_kernel(const float* __restrict__ e1,
                                const float* __restrict__ e2,
                                bf16* __restrict__ comb, int n, int C, int total)
{
    int tid = blockIdx.x * blockDim.x + threadIdx.x;
    if (tid >= total) return;
    int c = tid % C;
    int t = (tid / C) % (2 * n);
    int b = tid / (C * 2 * n);
    float v = (t < n) ? e1[((size_t)b * n + t) * C + c]
                      : e2[((size_t)b * n + (t - n)) * C + c];
    comb[tid] = (bf16)v;
}

// V[b,l,h*dh+d] -> Vt[((b*nh+h)*dh+d), l]  (per-head transpose for P*V frags)
__global__ void pin_vt_kernel(const bf16* __restrict__ V, bf16* __restrict__ Vt,
                              int L, int E, int nh, int total)
{
    int tid = blockIdx.x * blockDim.x + threadIdx.x;
    if (tid >= total) return;
    int dh = E / nh;
    int l  = tid % L;
    int t  = tid / L;
    int d  = t % dh;
    int bh = t / dh;
    int h  = bh % nh;
    int b  = bh / nh;
    Vt[tid] = V[((size_t)b * L + l) * E + h * dh + d];
}

// ---------------------------------------------------------------------------
// Flash attention: one wave per (b, head, 16-query tile), 64 keys/iteration
// (4 QK^T WMMAs + 4 PV WMMAs).  Softmax runs in the exp2 domain; p-values are
// written to LDS as they are produced so only the running stats stay live
// (no spills).  launch_bounds(128,1) gives the full VGPR budget.
// ---------------------------------------------------------------------------
__global__ void __launch_bounds__(128, 1)
pin_mha_kernel(const bf16* __restrict__ Q,
               const bf16* __restrict__ K,
               const bf16* __restrict__ Vt,
               bf16* __restrict__ O,
               int L, int E, int nh, int totalW)
{
    const int wave = threadIdx.x >> 5;
    const int lane = threadIdx.x & 31;
    const int gw = blockIdx.x * 4 + wave;
    if (gw >= totalW) return;

    const int dh = E / nh;                 // 32
    const int qTiles = L / 16;
    const int qt = gw % qTiles;
    const int bh = gw / qTiles;
    const int h  = bh % nh;
    const int b  = bh / nh;

    const int half  = lane >> 4;
    const int idx16 = lane & 15;
    const int kb    = half * 8;
    // softmax in exp2 domain: t = s * (1/sqrt(dh) * log2(e))
    const float cscale = 0.17677669529663687f * 1.44269504088896340736f;

    __shared__ bf16 Plds[4][16][64];
    bf16 (*P)[64] = Plds[wave];

    const bf16* qrow = Q + ((size_t)b * L + qt * 16 + idx16) * E + h * dh + kb;
    v16bf qa = ld2x8(qrow, qrow + 16);

    v8f o0 = {0.f,0.f,0.f,0.f,0.f,0.f,0.f,0.f};
    v8f o1 = o0;
    float mrow[8], lrow[8];
#pragma unroll
    for (int r = 0; r < 8; ++r) { mrow[r] = -3.0e38f; lrow[r] = 0.f; }

    const bf16* Kb  = K  + (size_t)b * L * E + h * dh + kb;
    const bf16* Vtb = Vt + ((size_t)(b * nh + h) * dh) * L;

    for (int kv = 0; kv < L; kv += 64) {
        v8f s0 = {0.f,0.f,0.f,0.f,0.f,0.f,0.f,0.f};
        v8f s1 = s0, s2 = s0, s3 = s0;
        const bf16* kr = Kb + (size_t)(kv + idx16) * E;
        { v16bf kf = ld2x8(kr, kr + 16); s0 = wmma_bf16(qa, kf, s0); }
        kr += (size_t)16 * E;
        { v16bf kf = ld2x8(kr, kr + 16); s1 = wmma_bf16(qa, kf, s1); }
        kr += (size_t)16 * E;
        { v16bf kf = ld2x8(kr, kr + 16); s2 = wmma_bf16(qa, kf, s2); }
        kr += (size_t)16 * E;
        { v16bf kf = ld2x8(kr, kr + 16); s3 = wmma_bf16(qa, kf, s3); }

        float rm[8];
#pragma unroll
        for (int r = 0; r < 8; ++r) {
            s0[r] *= cscale; s1[r] *= cscale; s2[r] *= cscale; s3[r] *= cscale;
            rm[r] = fmaxf(fmaxf(s0[r], s1[r]), fmaxf(s2[r], s3[r]));
        }
#pragma unroll
        for (int d = 1; d < 16; d <<= 1)
#pragma unroll
            for (int r = 0; r < 8; ++r) rm[r] = fmaxf(rm[r], __shfl_xor(rm[r], d, 32));

        // fused: exp2, stage to LDS, partial row-sum, rescale running state
        float rs[8];
#pragma unroll
        for (int r = 0; r < 8; ++r) {
            float mn = fmaxf(mrow[r], rm[r]);
            float alpha = exp2f(mrow[r] - mn);
            mrow[r] = mn;
            float p0 = exp2f(s0[r] - mn);
            float p1 = exp2f(s1[r] - mn);
            float p2 = exp2f(s2[r] - mn);
            float p3 = exp2f(s3[r] - mn);
            int row = half ? 8 + r : r;
            P[row][idx16]      = (bf16)p0;
            P[row][16 + idx16] = (bf16)p1;
            P[row][32 + idx16] = (bf16)p2;
            P[row][48 + idx16] = (bf16)p3;
            rs[r] = (p0 + p1) + (p2 + p3);
            lrow[r] *= alpha;
            o0[r] *= alpha;
            o1[r] *= alpha;
        }
#pragma unroll
        for (int d = 1; d < 16; d <<= 1)
#pragma unroll
            for (int r = 0; r < 8; ++r) rs[r] += __shfl_xor(rs[r], d, 32);
#pragma unroll
        for (int r = 0; r < 8; ++r) lrow[r] += rs[r];

        v16bf pa0 = ld2x8(&P[idx16][kb],      &P[idx16][16 + kb]);
        v16bf pa1 = ld2x8(&P[idx16][32 + kb], &P[idx16][48 + kb]);

        const bf16* vp0 = Vtb + (size_t)idx16 * L + kv + kb;
        { v16bf vf = ld2x8(vp0,      vp0 + 16); o0 = wmma_bf16(pa0, vf, o0); }
        { v16bf vf = ld2x8(vp0 + 32, vp0 + 48); o0 = wmma_bf16(pa1, vf, o0); }
        const bf16* vp1 = Vtb + (size_t)(16 + idx16) * L + kv + kb;
        { v16bf vf = ld2x8(vp1,      vp1 + 16); o1 = wmma_bf16(pa0, vf, o1); }
        { v16bf vf = ld2x8(vp1 + 32, vp1 + 48); o1 = wmma_bf16(pa1, vf, o1); }
    }

#pragma unroll
    for (int r = 0; r < 8; ++r) {
        float inv = 1.f / lrow[r];
        int row = qt * 16 + (half ? 8 + r : r);
        bf16* orow = O + ((size_t)b * L + row) * E + h * dh;
        orow[idx16]      = (bf16)(o0[r] * inv);
        orow[16 + idx16] = (bf16)(o1[r] * inv);
    }
}

__global__ void pin_gfused_kernel(const float* __restrict__ tau,
                                  const float* __restrict__ g1,
                                  const float* __restrict__ g2,
                                  float* __restrict__ gf, int C, int total)
{
    int tid = blockIdx.x * blockDim.x + threadIdx.x;
    if (tid >= total) return;
    int b = tid / C;
    float t = tau[b];
    gf[tid] = (1.f - t) * g1[tid] + t * g2[tid];
}

// A row = [ (1-t)*attn(p1) + t*attn(p2) (C) | gfused (G) ]
__global__ void pin_fuse_gather_kernel(const float* __restrict__ attn,
                                       const float* __restrict__ tau,
                                       const float* __restrict__ gfused,
                                       bf16* __restrict__ A,
                                       int n, int C, int G, int total)
{
    int tid = blockIdx.x * blockDim.x + threadIdx.x;
    if (tid >= total) return;
    int CG = C + G;
    int c = tid % CG;
    int i = (tid / CG) % n;
    int b = tid / (CG * n);
    float v;
    if (c < C) {
        float t = tau[b];
        float a1 = attn[((size_t)b * 2 * n + i)     * C + c];
        float a2 = attn[((size_t)b * 2 * n + n + i) * C + c];
        v = (1.f - t) * a1 + t * a2;
    } else {
        v = gfused[(size_t)b * G + (c - C)];
    }
    A[tid] = (bf16)v;
}

// gen row = [ fusionOut (C) | gfused (G) ]
__global__ void pin_gen_gather_kernel(const float* __restrict__ fo,
                                      const float* __restrict__ gfused,
                                      bf16* __restrict__ A,
                                      int n, int C, int G, int total)
{
    int tid = blockIdx.x * blockDim.x + threadIdx.x;
    if (tid >= total) return;
    int CG = C + G;
    int c = tid % CG;
    int i = (tid / CG) % n;
    int b = tid / (CG * n);
    float v = (c < C) ? fo[((size_t)b * n + i) * C + c]
                      : gfused[(size_t)b * G + (c - C)];
    A[tid] = (bf16)v;
}

__global__ void pin_final_kernel(const float* __restrict__ pc1,
                                 const float* __restrict__ tau,
                                 const float* __restrict__ delta,
                                 const float* __restrict__ feat,
                                 float* __restrict__ out, int n, int total)
{
    int tid = blockIdx.x * blockDim.x + threadIdx.x;
    if (tid >= total) return;
    int b = tid / n;
    float t = tau[b];
    const float* p = pc1 + (size_t)tid * 4;
    float* o = out + (size_t)tid * 4;
    o[0] = p[0] + t * delta[(size_t)tid * 3 + 0];
    o[1] = p[1] + t * delta[(size_t)tid * 3 + 1];
    o[2] = p[2] + t * delta[(size_t)tid * 3 + 2];
    o[3] = feat[tid];
}

// ---------------------------------------------------------------------------
// Host-side orchestration
// ---------------------------------------------------------------------------
static inline int nblk(long long total, int t = 256) { return (int)((total + t - 1) / t); }
static inline int al32(int x) { return (x + 31) & ~31; }
static inline int al16(int x) { return (x + 15) & ~15; }

struct PinLayer { int wi; int K; int N; };
// wi = leaf index relative to d_in[3] (dict insertion order: w then b per layer)
static const PinLayer kLayers[27] = {
    { 0,   4,  64}, { 2,  64,  64}, { 4,  64, 128},   // sa1
    { 6, 131, 128}, { 8, 128, 128}, {10, 128, 256},   // sa2
    {12, 259, 256}, {14, 256, 256}, {16, 256, 512},   // sa3
    {18, 768, 256}, {20, 256, 256},                   // fp3
    {22, 384, 256}, {24, 256, 128},                   // fp2
    {26, 132, 128}, {28, 128, 128},                   // fp1
    {30, 128, 128}, {32, 128, 128}, {34, 128, 128}, {36, 128, 128}, // attn q,k,v,o
    {38, 640, 256}, {40, 256, 128},                   // fusion
    {42, 640, 256}, {44, 256, 128}, {46, 128,   3},   // coord
    {48, 640, 256}, {50, 256, 128}, {52, 128,   1},   // featm
};

extern "C" void kernel_launch(void* const* d_in, const int* in_sizes, int n_in,
                              void* d_out, int out_size, void* d_ws, size_t ws_size,
                              hipStream_t stream)
{
    const int B = 4, N = 2048;

    const float* pc1 = (const float*)d_in[0];
    const float* pc2 = (const float*)d_in[1];
    const float* tau = (const float*)d_in[2];
    const float* Pp[60];
    for (int i = 0; i < 60; ++i) Pp[i] = (i + 3 < n_in) ? (const float*)d_in[i + 3] : nullptr;
    (void)in_sizes; (void)out_size; (void)ws_size;

    size_t off = 0;
    auto alloc = [&](size_t bytes) -> void* {
        void* p = (char*)d_ws + off;
        off += (bytes + 255) & ~(size_t)255;
        return p;
    };

    float* pts1  = (float*)alloc((size_t)B*N*128*4);
    float* pts2  = (float*)alloc((size_t)B*N*128*4);
    float* g1    = (float*)alloc((size_t)B*512*4);
    float* g2    = (float*)alloc((size_t)B*512*4);
    int*   idxb  = (int*)  alloc((size_t)B*512*4);
    int*   gidx  = (int*)  alloc((size_t)B*512*32*4);
    float* l1x   = (float*)alloc((size_t)B*512*3*4);
    float* l2x   = (float*)alloc((size_t)B*128*3*4);
    float* l3x   = (float*)alloc((size_t)B*3*4);
    float* l1p   = (float*)alloc((size_t)B*512*128*4);
    float* l2p   = (float*)alloc((size_t)B*128*256*4);
    float* l2pn  = (float*)alloc((size_t)B*128*256*4);
    float* l1pn  = (float*)alloc((size_t)B*512*128*4);
    bf16*  P0    = (bf16*) alloc((size_t)12*1024*1024);   // max bf16 A buffer
    bf16*  P1    = (bf16*) alloc((size_t)9*1024*1024);
    float* F0    = (float*)alloc((size_t)65536*128*4);    // f32 scratch (32MB)
    bf16*  comb  = (bf16*) alloc((size_t)B*2*N*128*2);
    bf16*  qb    = (bf16*) alloc((size_t)B*2*N*128*2);
    bf16*  kb_   = (bf16*) alloc((size_t)B*2*N*128*2);
    bf16*  vb    = (bf16*) alloc((size_t)B*2*N*128*2);
    bf16*  vt    = (bf16*) alloc((size_t)B*2*N*128*2);
    bf16*  aO    = (bf16*) alloc((size_t)B*2*N*128*2);
    float* attnO = (float*)alloc((size_t)B*2*N*128*4);
    float* gfus  = (float*)alloc((size_t)B*512*4);
    bf16*  GenA  = (bf16*) alloc((size_t)B*N*640*2);
    float* fusO  = (float*)alloc((size_t)B*N*128*4);
    bf16*  C2bf  = (bf16*) alloc((size_t)B*N*128*2);
    float* delta = (float*)alloc((size_t)B*N*3*4);
    float* featv = (float*)alloc((size_t)B*N*4);

    // ---- Pack all layer weights into B-fragment order (bf16, zero padded)
    bf16* Wp[27];
    for (int li = 0; li < 27; ++li) {
        int Kp = al32(kLayers[li].K), Np = al16(kLayers[li].N);
        long long tot = (long long)(Kp / 32) * Np * 32;
        Wp[li] = (bf16*)alloc((size_t)tot * 2);
        pin_packw_kernel<<<nblk(tot), 256, 0, stream>>>(
            Pp[kLayers[li].wi], Wp[li], kLayers[li].K, kLayers[li].N, Np, (int)tot);
    }

    auto G = [&](const bf16* A, int li, void* C, int ldc, int M, int relu, int obf) {
        int Kp = al32(kLayers[li].K);
        int Nn = kLayers[li].N;
        int Np = al16(Nn);
        if ((Np & 63) == 0) {
            // MT=2, NT=4: block of 4 waves covers 128 rows x 64 cols
            dim3 g(Np / 64, (M + 127) / 128);
            pin_gemmT_kernel<2, 4><<<g, 128, 0, stream>>>(
                A, Kp, Wp[li], Pp[kLayers[li].wi + 1], C, ldc, M, Nn, Np, Kp, relu, obf);
        } else {
            // Narrow heads (Npad == 16): MT=1, NT=1
            dim3 g(1, (M + 63) / 64);
            pin_gemmT_kernel<1, 1><<<g, 128, 0, stream>>>(
                A, Kp, Wp[li], Pp[kLayers[li].wi + 1], C, ldc, M, Nn, Np, Kp, relu, obf);
        }
    };

    auto encode = [&](const float* pc, float* pts_out, float* g_out) {
        // ---- SA1: 2048 -> 512, r=0.2, ns=32, MLP 4->64->64->128
        pin_fps_kernel<<<B, FPS_T, 0, stream>>>(pc, 4, N, 512, idxb);
        pin_gather_xyz_kernel<<<nblk(B*512), 256, 0, stream>>>(pc, 4, idxb, l1x, 512, N, B*512);
        pin_ball_query_kernel<<<nblk(B*512), 256, 0, stream>>>(pc, 4, l1x, N, 512, 0.04f, 32, gidx, B*512);
        pin_sa_gather_kernel<<<nblk((long long)B*512*32), 256, 0, stream>>>(
            pc, 4, pc + 3, 4, l1x, gidx, P0, 512, 32, N, 1, 32, B*512*32);
        G(P0, 0, P1, 64,  B*512*32, 1, 1);
        G(P1, 1, P0, 64,  B*512*32, 1, 1);
        G(P0, 2, F0, 128, B*512*32, 1, 0);
        pin_maxpool_kernel<<<nblk((long long)B*512*128), 256, 0, stream>>>(F0, l1p, B*512, 32, 128, B*512*128);

        // ---- SA2: 512 -> 128, r=0.4, ns=64, MLP 131->128->128->256
        pin_fps_kernel<<<B, FPS_T, 0, stream>>>(l1x, 3, 512, 128, idxb);
        pin_gather_xyz_kernel<<<nblk(B*128), 256, 0, stream>>>(l1x, 3, idxb, l2x, 128, 512, B*128);
        pin_ball_query_kernel<<<nblk(B*128), 256, 0, stream>>>(l1x, 3, l2x, 512, 128, 0.16f, 64, gidx, B*128);
        pin_sa_gather_kernel<<<nblk((long long)B*128*64), 256, 0, stream>>>(
            l1x, 3, l1p, 128, l2x, gidx, P0, 128, 64, 512, 128, 160, B*128*64);
        G(P0, 3, P1, 128, B*128*64, 1, 1);
        G(P1, 4, P0, 128, B*128*64, 1, 1);
        G(P0, 5, F0, 256, B*128*64, 1, 0);
        pin_maxpool_kernel<<<nblk((long long)B*128*256), 256, 0, stream>>>(F0, l2p, B*128, 64, 256, B*128*256);

        // ---- SA3: 128 -> 1, r=0.8, ns=128, MLP 259->256->256->512
        pin_fps_kernel<<<B, FPS_T, 0, stream>>>(l2x, 3, 128, 1, idxb);
        pin_gather_xyz_kernel<<<nblk(B), 256, 0, stream>>>(l2x, 3, idxb, l3x, 1, 128, B);
        pin_ball_query_kernel<<<nblk(B), 256, 0, stream>>>(l2x, 3, l3x, 128, 1, 0.64f, 128, gidx, B);
        pin_sa_gather_kernel<<<nblk((long long)B*128), 256, 0, stream>>>(
            l2x, 3, l2p, 256, l3x, gidx, P0, 1, 128, 128, 256, 288, B*128);
        G(P0, 6, P1, 256, B*128, 1, 1);
        G(P1, 7, P0, 256, B*128, 1, 1);
        G(P0, 8, F0, 512, B*128, 1, 0);
        pin_maxpool_kernel<<<nblk((long long)B*512), 256, 0, stream>>>(F0, g_out, B, 128, 512, B*512);

        // ---- FP3 (broadcast, m=1): [l2p(256) | gf(512)] -> 256 -> 256
        pin_fp_concat_kernel<<<nblk(B*128), 256, 0, stream>>>(
            l2x, 3, l3x, l2p, 256, 256, g_out, 512, P0, 768, 128, 1, B*128);
        G(P0, 9,  P1, 256, B*128, 1, 1);
        G(P1, 10, l2pn, 256, B*128, 1, 0);

        // ---- FP2: [l1p(128) | interp3(l2pn,256)] -> 256 -> 128
        pin_fp_concat_kernel<<<nblk(B*512), 256, 0, stream>>>(
            l1x, 3, l2x, l1p, 128, 128, l2pn, 256, P0, 384, 512, 128, B*512);
        G(P0, 11, P1, 256, B*512, 1, 1);
        G(P1, 12, l1pn, 128, B*512, 1, 0);

        // ---- FP1: [pc(4) | interp3(l1pn,128)] -> 128 -> 128
        pin_fp_concat_kernel<<<nblk(B*N), 256, 0, stream>>>(
            pc, 4, l1x, pc, 4, 4, l1pn, 128, P0, 160, N, 512, B*N);
        G(P0, 13, P1, 128, B*N, 1, 1);
        G(P1, 14, pts_out, 128, B*N, 1, 0);
    };

    encode(pc1, pts1, g1);
    encode(pc2, pts2, g2);

    // ---- Attention over comb = [enc1 | enc2] tokens (L=4096, E=128, 4 heads)
    const int L = 2 * N, E = 128, NH = 4;
    pin_comb_kernel<<<nblk((long long)B*L*E), 256, 0, stream>>>(pts1, pts2, comb, N, E, B*L*E);
    G(comb, 15, qb,  E, B*L, 0, 1);
    G(comb, 16, kb_, E, B*L, 0, 1);
    G(comb, 17, vb,  E, B*L, 0, 1);
    pin_vt_kernel<<<nblk((long long)B*L*E), 256, 0, stream>>>(vb, vt, L, E, NH, B*L*E);
    {
        int totalW = B * NH * (L / 16);
        pin_mha_kernel<<<(totalW + 3) / 4, 128, 0, stream>>>(qb, kb_, vt, aO, L, E, NH, totalW);
    }
    G(aO, 18, attnO, E, B*L, 0, 0);   // output projection

    // ---- Fusion + heads
    pin_gfused_kernel<<<nblk(B*512), 256, 0, stream>>>(tau, g1, g2, gfus, 512, B*512);
    pin_fuse_gather_kernel<<<nblk((long long)B*N*640), 256, 0, stream>>>(
        attnO, tau, gfus, P0, N, 128, 512, B*N*640);
    G(P0, 19, P1,   256, B*N, 1, 1);
    G(P1, 20, fusO, 128, B*N, 1, 0);
    pin_gen_gather_kernel<<<nblk((long long)B*N*640), 256, 0, stream>>>(
        fusO, gfus, GenA, N, 128, 512, B*N*640);

    // coord: 640 -> 256 -> 128 -> 3 (no relu on last)
    G(GenA, 21, P1,    256, B*N, 1, 1);
    G(P1,   22, C2bf,  128, B*N, 1, 1);
    G(C2bf, 23, delta, 3,   B*N, 0, 0);
    // featm: 640 -> 256 -> 128 -> 1 (no relu on last)
    G(GenA, 24, P1,    256, B*N, 1, 1);
    G(P1,   25, C2bf,  128, B*N, 1, 1);
    G(C2bf, 26, featv, 1,   B*N, 0, 0);

    pin_final_kernel<<<nblk(B*N), 256, 0, stream>>>(pc1, tau, delta, featv,
                                                    (float*)d_out, N, B*N);
}
